// DAttention_16844861735478
// MI455X (gfx1250) — matrix operand order repaired and checked by reading, not compiled, not run on hardware
//
#include <hip/hip_runtime.h>

#define B_   8
#define C_   768
#define H_   28
#define W_   28
#define G_   6
#define GC_  128
#define NH_  12
#define HC_  64
#define NS_  784          // H*W
#define BH_  (B_*NH_)     // 96
#define M_   (B_*NS_)     // 6272 rows for the projection GEMMs
#define LDV_ 800          // padded K for P@V (784 -> 800, zero tail)
#define SCALE_ 0.125f     // HC^-0.5
#define EPS_ 1e-5f

typedef __attribute__((ext_vector_type(16))) _Float16 v16h;
typedef __attribute__((ext_vector_type(8)))  float    v8f;
typedef __attribute__((ext_vector_type(4)))  float    vf4;
typedef __attribute__((ext_vector_type(2)))  float    vf2;

#define V8F_ZERO {0.f,0.f,0.f,0.f,0.f,0.f,0.f,0.f}

union H16 { v16h v; vf4 f4[2]; _Float16 h[16]; };
union H8  { vf4 f4;            _Float16 h[8];  };

__device__ __forceinline__ v8f wmma_f16(v16h a, v16h b, v8f c) {
  // D = A(16x32 f16) * B(32x16 f16) + C(16x16 f32)
  return __builtin_amdgcn_wmma_f32_16x16x32_f16(
      /*neg_a=*/false, a, /*neg_b=*/false, b,
      /*c_mod=*/(short)0, c, /*reuse_a=*/false, /*reuse_b=*/false);
}

// A fragment from row-major f16 [rows][ld]: lane l -> row (l&15),
// K chunks of 8 halves at k0+(l>>4)*8 and +16  (ISA 7.12.2, 16-bit A 16x32)
__device__ __forceinline__ v16h frag_a_f16(const _Float16* A, int row0, int ld, int k0) {
  int lane = threadIdx.x & 31;
  const _Float16* p = A + (size_t)(row0 + (lane & 15)) * ld + k0 + ((lane >> 4) << 3);
  H16 u;
  u.f4[0] = *(const vf4*)p;
  u.f4[1] = *(const vf4*)(p + 16);
  return u.v;
}

// B fragment: lane l -> column (l&15), K chunks of 8 halves at k0+(l>>4)*8 and +16.
// Works directly on row-major W[N][K] (weights) / K-rows / transposed V.
__device__ __forceinline__ v16h frag_b_f16(const _Float16* Bm, int col0, int ld, int k0) {
  int lane = threadIdx.x & 31;
  const _Float16* p = Bm + (size_t)(col0 + (lane & 15)) * ld + k0 + ((lane >> 4) << 3);
  H16 u;
  u.f4[0] = *(const vf4*)p;
  u.f4[1] = *(const vf4*)(p + 16);
  return u.v;
}

// ---------------------------------------------------------------------------
// Projection GEMM: D[M x N] = A[M x K](f16) @ W[N x K]^T(f16) + bias
// One wave per block, 32(M) x 64(N) per wave: each B fragment feeds 2 WMMAs.
// MODE 0: store f16 row-major [M][N]
// MODE 1: store f32 row-major [M][N]
// MODE 2: store f16 transposed per-batch [b][N][ldT] (for V), b from 16-row half
// ---------------------------------------------------------------------------
template<int MODE>
__device__ __forceinline__ void gemm_store(const v8f* acc, int row0, int nb,
                                           const float* __restrict__ bias,
                                           void* __restrict__ outp, int N, int ldT,
                                           int nl, int hi) {
#pragma unroll
  for (int j = 0; j < 4; ++j) {
    int n = nb * 64 + j * 16 + nl;
    float bv = bias[n];
    if (MODE == 0) {
      _Float16* o = (_Float16*)outp;
#pragma unroll
      for (int r = 0; r < 8; ++r) {
        int m = row0 + hi * 8 + r;                  // D layout: lane col nl, row 8*hi+r
        o[(size_t)m * N + n] = (_Float16)(acc[j][r] + bv);
      }
    } else if (MODE == 1) {
      float* o = (float*)outp;
#pragma unroll
      for (int r = 0; r < 8; ++r) {
        int m = row0 + hi * 8 + r;
        o[(size_t)m * N + n] = acc[j][r] + bv;
      }
    } else {
      // transposed store: per lane 8 contiguous halves -> one b128 store
      _Float16* o = (_Float16*)outp;
      int bidx = row0 / NS_;                        // 16-row halves never straddle batches
      int s0   = row0 - bidx * NS_;
      H8 u;
#pragma unroll
      for (int r = 0; r < 8; ++r) u.h[r] = (_Float16)(acc[j][r] + bv);
      *(vf4*)(o + ((size_t)bidx * C_ + n) * ldT + s0 + hi * 8) = u.f4;
    }
  }
}

template<int MODE>
__global__ void gemm_proj(const _Float16* __restrict__ A, const _Float16* __restrict__ Wt,
                          const float* __restrict__ bias, void* __restrict__ outp,
                          int N, int K, int ldT) {
  const int mt   = blockIdx.x;          // 32-row tile
  const int nb   = blockIdx.y;          // 64-col block
  const int lane = threadIdx.x & 31;
  const int nl = lane & 15, hi = lane >> 4;

  v8f acc0[4] = {V8F_ZERO, V8F_ZERO, V8F_ZERO, V8F_ZERO};
  v8f acc1[4] = {V8F_ZERO, V8F_ZERO, V8F_ZERO, V8F_ZERO};
  for (int k0 = 0; k0 < K; k0 += 32) {
    v16h a0 = frag_a_f16(A, mt * 32,      K, k0);
    v16h a1 = frag_a_f16(A, mt * 32 + 16, K, k0);
    if (k0 + 32 < K) {                  // hint next A strip into cache (global_prefetch)
      __builtin_prefetch(A + (size_t)(mt * 32 + (lane & 15)) * K + k0 + 32, 0, 0);
    }
#pragma unroll
    for (int j = 0; j < 4; ++j) {
      v16h b = frag_b_f16(Wt, nb * 64 + j * 16, K, k0);
      acc0[j] = wmma_f16(a0, b, acc0[j]);
      acc1[j] = wmma_f16(a1, b, acc1[j]);
    }
  }
  gemm_store<MODE>(acc0, mt * 32,      nb, bias, outp, N, ldT, nl, hi);
  gemm_store<MODE>(acc1, mt * 32 + 16, nb, bias, outp, N, ldT, nl, hi);
}

// ---------------------------------------------------------------------------
// Fused attention: scores (WMMA) -> LDS -> softmax (LDS) -> attn write (once)
// -> P@V (WMMA, A-fragments converted f32->f16 straight out of LDS).
// One block = 2 waves per (bh, 16-query-row tile). LDS strip: 16 x 800 f32.
// ---------------------------------------------------------------------------
__global__ void attn_fused(const _Float16* __restrict__ qh, const _Float16* __restrict__ kh,
                           const _Float16* __restrict__ vth,
                           float* __restrict__ attn, _Float16* __restrict__ oinh) {
  __shared__ float sc[16][LDV_];        // 51.2 KB of the 320 KB/WGP LDS
  __shared__ float red[16][4];

  const int mt = blockIdx.x;            // 0..48
  const int bh = blockIdx.y;            // 0..95
  const int b = bh / NH_, nh = bh % NH_;
  const int t = threadIdx.x;            // 0..63
  const int w = t >> 5;                 // wave id (0/1)
  const int lane = t & 31;
  const int nl = lane & 15, hi = lane >> 4;

  // zero the K pad (cols 784..799) so the P@V loop runs exactly 25 x 32
  for (int idx = t; idx < 16 * 16; idx += 64)
    sc[idx >> 4][NS_ + (idx & 15)] = 0.f;

  // ---- phase 1: scores -> LDS (wave0: n-tiles 0..24, wave1: 25..48) ----
  const _Float16* Aq = qh + (size_t)b * NS_ * C_ + nh * HC_;
  const _Float16* Bk = kh + (size_t)b * NS_ * C_ + nh * HC_;
  v16h a0 = frag_a_f16(Aq, mt * 16, C_, 0);
  v16h a1 = frag_a_f16(Aq, mt * 16, C_, 32);
  const int nt0 = w ? 25 : 0, nt1 = w ? 49 : 25;
  for (int nt = nt0; nt < nt1; ++nt) {
    v8f acc = V8F_ZERO;
    v16h b0 = frag_b_f16(Bk, nt * 16, C_, 0);
    v16h b1 = frag_b_f16(Bk, nt * 16, C_, 32);
    acc = wmma_f16(a0, b0, acc);
    acc = wmma_f16(a1, b1, acc);
#pragma unroll
    for (int r = 0; r < 8; ++r)
      sc[hi * 8 + r][nt * 16 + nl] = acc[r] * SCALE_;
  }
  __syncthreads();

  // ---- phase 2: softmax over 784 in LDS, write normalized attn to d_out ----
  {
    const int row = t >> 2, sub = t & 3;        // 4 threads per row (same wave)
    float* prow = &sc[row][0];
    float pmax = -3.4e38f;
#pragma unroll 4
    for (int j = 0; j < 196; ++j) pmax = fmaxf(pmax, prow[sub + 4 * j]);
    red[row][sub] = pmax;
    __syncthreads();
    float mx = fmaxf(fmaxf(red[row][0], red[row][1]), fmaxf(red[row][2], red[row][3]));
    float psum = 0.f;
#pragma unroll 4
    for (int j = 0; j < 196; ++j) {
      float e = expf(prow[sub + 4 * j] - mx);
      prow[sub + 4 * j] = e;
      psum += e;
    }
    __syncthreads();
    red[row][sub] = psum;
    __syncthreads();
    float inv = 1.f / (red[row][0] + red[row][1] + red[row][2] + red[row][3]);
    float* gout = attn + ((size_t)bh * NS_ + mt * 16 + row) * NS_;
#pragma unroll 4
    for (int j = 0; j < 196; ++j) {
      float pv = prow[sub + 4 * j] * inv;
      prow[sub + 4 * j] = pv;
      gout[sub + 4 * j] = pv;
    }
  }
  __syncthreads();

  // ---- phase 3: P @ V (wave0: d-tiles 0,1; wave1: 2,3) ----
  const _Float16* V = vth + ((size_t)b * C_ + nh * HC_) * LDV_;
  v8f acc2[2] = {V8F_ZERO, V8F_ZERO};
  for (int ks = 0; ks < 25; ++ks) {
    int k0 = ks * 32;
    // A fragment from LDS f32 probs, converted to f16 in-register
    const float* pr = &sc[nl][k0 + hi * 8];
    vf4 q0 = *(const vf4*)pr;
    vf4 q1 = *(const vf4*)(pr + 4);
    vf4 q2 = *(const vf4*)(pr + 16);
    vf4 q3 = *(const vf4*)(pr + 20);
    H16 u;
#pragma unroll
    for (int i = 0; i < 4; ++i) {
      u.h[i]      = (_Float16)q0[i];
      u.h[4 + i]  = (_Float16)q1[i];
      u.h[8 + i]  = (_Float16)q2[i];
      u.h[12 + i] = (_Float16)q3[i];
    }
#pragma unroll
    for (int jj = 0; jj < 2; ++jj) {
      int j = w * 2 + jj;
      v16h bb = frag_b_f16(V, j * 16, LDV_, k0);
      acc2[jj] = wmma_f16(u.v, bb, acc2[jj]);
    }
  }
#pragma unroll
  for (int jj = 0; jj < 2; ++jj) {
    int j = w * 2 + jj;
    int c = nh * HC_ + j * 16 + nl;
#pragma unroll
    for (int r = 0; r < 8; ++r) {
      int s = mt * 16 + hi * 8 + r;
      oinh[((size_t)b * NS_ + s) * C_ + c] = (_Float16)acc2[jj][r];
    }
  }
}

// ---------------------------------------------------------------------------
// Depthwise 5x5 conv + LayerNorm(128ch) + exact GELU + offset proj + tanh ->
// bilinear sampling coords. One 128-thread block per (b*G+g, spatial) point.
// ---------------------------------------------------------------------------
__global__ void dwconv_offset(const _Float16* __restrict__ qh, const float* __restrict__ dw_w,
                              const float* __restrict__ dw_b, const float* __restrict__ ln_g,
                              const float* __restrict__ ln_b, const float* __restrict__ off_w,
                              vf2* __restrict__ coords) {
  const int s = blockIdx.x, bg = blockIdx.y;
  const int b = bg / G_, g = bg % G_;
  const int gc = threadIdx.x;           // 0..127
  const int h = s / W_, w = s % W_;
  const int c = g * GC_ + gc;

  float acc = dw_b[gc];
#pragma unroll
  for (int dy = 0; dy < 5; ++dy) {
    int y = h + dy - 2;
    if (y < 0 || y >= H_) continue;
#pragma unroll
    for (int dx = 0; dx < 5; ++dx) {
      int x = w + dx - 2;
      if (x < 0 || x >= W_) continue;
      acc += (float)qh[((size_t)b * NS_ + y * W_ + x) * C_ + c] * dw_w[gc * 25 + dy * 5 + dx];
    }
  }
  __shared__ float r1[128], r2[128];
  r1[gc] = acc; r2[gc] = acc * acc;
  __syncthreads();
  for (int st = 64; st > 0; st >>= 1) {
    if (gc < st) { r1[gc] += r1[gc + st]; r2[gc] += r2[gc + st]; }
    __syncthreads();
  }
  float mu  = r1[0] * (1.f / GC_);
  float var = r2[0] * (1.f / GC_) - mu * mu;
  __syncthreads();
  float tn = (acc - mu) * rsqrtf(var + EPS_) * ln_g[gc] + ln_b[gc];
  tn = 0.5f * tn * (1.f + erff(tn * 0.70710678118654752f));   // exact GELU
  r1[gc] = off_w[gc] * tn;          // off_w[0][gc] -> y channel
  r2[gc] = off_w[GC_ + gc] * tn;    // off_w[1][gc] -> x channel
  __syncthreads();
  for (int st = 64; st > 0; st >>= 1) {
    if (gc < st) { r1[gc] += r1[gc + st]; r2[gc] += r2[gc + st]; }
    __syncthreads();
  }
  if (gc == 0) {
    float oy = tanhf(r1[0]) * (2.0f / H_);    // off_range * OFR = (1/H)*2
    float ox = tanhf(r2[0]) * (2.0f / W_);
    float py = oy + ((h + 0.5f) / H_) * 2.f - 1.f;
    float px = ox + ((w + 0.5f) / W_) * 2.f - 1.f;
    vf2 cc;
    cc.x = (px + 1.f) * 0.5f * (W_ - 1);      // grid = pos[..., ::-1] -> x first
    cc.y = (py + 1.f) * 0.5f * (H_ - 1);
    coords[(size_t)bg * NS_ + s] = cc;
  }
}

// Bilinear grid-sample of x2 (f32 [b,s,c]) -> xsh (f16 [b,s,c])
__global__ void sample_x2k(const float* __restrict__ x2, const vf2* __restrict__ coords,
                           _Float16* __restrict__ xsh) {
  const int s = blockIdx.x, b = blockIdx.y;
  for (int c = threadIdx.x; c < C_; c += blockDim.x) {
    int g = c >> 7;
    vf2 pc = coords[((size_t)(b * G_ + g)) * NS_ + s];
    float x = pc.x, y = pc.y;
    float xf = floorf(x), yf = floorf(y);
    float wx1 = x - xf, wy1 = y - yf;
    int xi = (int)xf, yi = (int)yf;
    float accv = 0.f;
#pragma unroll
    for (int dy = 0; dy < 2; ++dy) {
#pragma unroll
      for (int dx = 0; dx < 2; ++dx) {
        int yy = yi + dy, xx = xi + dx;
        if (xx >= 0 && xx < W_ && yy >= 0 && yy < H_) {
          float wgt = (dy ? wy1 : 1.f - wy1) * (dx ? wx1 : 1.f - wx1);
          accv += x2[((size_t)b * NS_ + yy * W_ + xx) * C_ + c] * wgt;
        }
      }
    }
    xsh[((size_t)b * NS_ + s) * C_ + c] = (_Float16)accv;
  }
}

__global__ void cvt_f32_f16(const float* __restrict__ src, _Float16* __restrict__ dst, int n) {
  int i = blockIdx.x * blockDim.x + threadIdx.x;
  int stride = gridDim.x * blockDim.x;
  for (; i < n; i += stride) dst[i] = (_Float16)src[i];
}
__global__ void zero_f16(_Float16* __restrict__ dst, int n) {
  int i = blockIdx.x * blockDim.x + threadIdx.x;
  int stride = gridDim.x * blockDim.x;
  for (; i < n; i += stride) dst[i] = (_Float16)0.f;
}

extern "C" void kernel_launch(void* const* d_in, const int* in_sizes, int n_in,
                              void* d_out, int out_size, void* d_ws, size_t ws_size,
                              hipStream_t stream) {
  (void)in_sizes; (void)n_in; (void)out_size; (void)ws_size;
  const float* x1   = (const float*)d_in[0];
  const float* x2   = (const float*)d_in[1];
  const float* dw_w = (const float*)d_in[2];
  const float* dw_b = (const float*)d_in[3];
  const float* ln_g = (const float*)d_in[4];
  const float* ln_b = (const float*)d_in[5];
  const float* offw = (const float*)d_in[6];
  const float* wq   = (const float*)d_in[7];
  const float* bq   = (const float*)d_in[8];
  const float* wk   = (const float*)d_in[9];
  const float* bk   = (const float*)d_in[10];
  const float* wv   = (const float*)d_in[11];
  const float* bv   = (const float*)d_in[12];
  const float* wo   = (const float*)d_in[13];
  const float* bo   = (const float*)d_in[14];

  float* outp = (float*)d_out;                       // [B][NS][C] f32
  float* attn = outp + (size_t)M_ * C_;              // [B*NH][NS][NS] f32

  char* ws = (char*)d_ws;
  auto take = [&](size_t bytes) -> void* {
    void* p = (void*)ws;
    ws += (bytes + 255) & ~(size_t)255;
    return p;
  };
  _Float16* x1h  = (_Float16*)take((size_t)M_ * C_ * 2);
  _Float16* qh   = (_Float16*)take((size_t)M_ * C_ * 2);
  _Float16* xsh  = (_Float16*)take((size_t)M_ * C_ * 2);
  _Float16* kh   = (_Float16*)take((size_t)M_ * C_ * 2);
  _Float16* oinh = (_Float16*)take((size_t)M_ * C_ * 2);
  _Float16* vth  = (_Float16*)take((size_t)B_ * C_ * LDV_ * 2);
  _Float16* wqh  = (_Float16*)take((size_t)C_ * C_ * 2);
  _Float16* wkh  = (_Float16*)take((size_t)C_ * C_ * 2);
  _Float16* wvh  = (_Float16*)take((size_t)C_ * C_ * 2);
  _Float16* woh  = (_Float16*)take((size_t)C_ * C_ * 2);
  vf2*      crd  = (vf2*)take((size_t)B_ * G_ * NS_ * sizeof(vf2));

  const dim3 b32(32), b64(64), b128(128), b256(256);
  const dim3 ggemm(M_ / 32, C_ / 64);                // 196 x 12 waves

  // f32 -> f16 packing (activations + weights)
  cvt_f32_f16<<<512, b256, 0, stream>>>(x1, x1h, M_ * C_);
  cvt_f32_f16<<<256, b256, 0, stream>>>(wq, wqh, C_ * C_);
  cvt_f32_f16<<<256, b256, 0, stream>>>(wk, wkh, C_ * C_);
  cvt_f32_f16<<<256, b256, 0, stream>>>(wv, wvh, C_ * C_);
  cvt_f32_f16<<<256, b256, 0, stream>>>(wo, woh, C_ * C_);

  // q = x1 @ Wq^T + bq  (f16 [b,s,c])
  gemm_proj<0><<<ggemm, b32, 0, stream>>>(x1h, wqh, bq, qh, C_, C_, 0);

  // offset path -> sampling coordinates
  dwconv_offset<<<dim3(NS_, B_ * G_), b128, 0, stream>>>(qh, dw_w, dw_b, ln_g, ln_b, offw, crd);

  // grid-sample x2 -> xs (f16 [b,s,c])
  sample_x2k<<<dim3(NS_, B_), b256, 0, stream>>>(x2, crd, xsh);

  // k = xs @ Wk^T + bk  (f16 [b,s,c])
  gemm_proj<0><<<ggemm, b32, 0, stream>>>(xsh, wkh, bk, kh, C_, C_, 0);

  // v = xs @ Wv^T + bv, stored transposed f16 [b][c][800] with zero pad
  zero_f16<<<512, b256, 0, stream>>>(vth, B_ * C_ * LDV_);
  gemm_proj<2><<<ggemm, b32, 0, stream>>>(xsh, wvh, bv, vth, C_, C_, LDV_);

  // fused attention: scores -> softmax -> attn(d_out) -> P@V -> oinh
  attn_fused<<<dim3(49, BH_), b64, 0, stream>>>(qh, kh, vth, attn, oinh);

  // out = oin @ Wo^T + bo  (f32 [b,s,c] -> d_out)
  gemm_proj<1><<<ggemm, b32, 0, stream>>>(oinh, woh, bo, outp, C_, C_, 0);
}
